// DenseSAKELayer_13108240187514
// MI455X (gfx1250) — matrix operand order repaired and checked by reading, not compiled, not run on hardware
//
#include <hip/hip_runtime.h>
#include <hip/hip_bf16.h>
#include <stdint.h>

// ---------------- constants ----------------
#define NN 512   // nodes
#define FD 64    // feature dim
#define HD 64    // hidden dim
#define CD 32    // edge channels
#define KR 50    // rbf dim (padded to 64 for WMMA K)

typedef __attribute__((ext_vector_type(16))) _Float16 v16h;
typedef __attribute__((ext_vector_type(8)))  float    v8f;

union H16 { v16h v; _Float16 h[16]; uint32_t u[8]; };
union HU  { _Float16 h; uint16_t u; };

__device__ __forceinline__ v8f wmma_f16(v16h a, v16h b, v8f c) {
  // D = A(16x32 f16) * B(32x16 f16) + C(16x16 f32)
  return __builtin_amdgcn_wmma_f32_16x16x32_f16(false, a, false, b, (short)0, c,
                                                false, false);
}

// B fragment: packed layout [frag][lane][8 dwords] (two f16 per dword)
__device__ __forceinline__ v16h load_bfrag(const uint32_t* __restrict__ pk, int f, int lane) {
  const uint32_t* p = pk + ((f * 32 + lane) << 3);
  H16 r;
#pragma unroll
  for (int w = 0; w < 8; ++w) r.u[w] = p[w];
  return r.v;
}

// A-matrix (16x32, f16) per-lane K offset for element e (0..15):
// lanes 0-15 cover K {0..7, 16..23}; lanes 16-31 cover K {8..15, 24..31}
__device__ __forceinline__ int a_koff(int e, int b0) {
  return (e < 8) ? (b0 + e) : (b0 + 8 + e);
}

__device__ __forceinline__ float siluf(float v) { return v / (1.f + expf(-v)); }

// ---------------- kernel A: per-node projections ----------------
// hW1b[n,k] = h[n]@W_df_in[0:64,k] + b_df_in[k]   (k<50, else 0)
// hW2 [n,k] = h[n]@W_df_in[64:128,k]              (k<50, else 0)
// ewj[n,c]  = h[n]@W_ew[0:64,c]   ewi[n,c] = h[n]@W_ew[64:128,c]
// hsj[n]    = h[n]@W_sa[0:64]     hsi[n]   = h[n]@W_sa[64:128]
__global__ void sake_prep(const float* __restrict__ h,
                          const float* __restrict__ W_df_in,
                          const float* __restrict__ b_df_in,
                          const float* __restrict__ W_sa,
                          const float* __restrict__ W_ew,
                          float* __restrict__ hW1b, float* __restrict__ hW2,
                          float* __restrict__ hsj,  float* __restrict__ hsi,
                          float* __restrict__ ewj,  float* __restrict__ ewi) {
  const int n = blockIdx.x;
  const int t = threadIdx.x;  // 64 threads
  __shared__ float hn[FD];
  hn[t] = h[n * FD + t];
  __syncthreads();

  float a1 = 0.f, a2 = 0.f;
  if (t < KR) {
    for (int f = 0; f < FD; ++f) {
      a1 += hn[f] * W_df_in[f * KR + t];
      a2 += hn[f] * W_df_in[(FD + f) * KR + t];
    }
    a1 += b_df_in[t];
  }
  hW1b[n * 64 + t] = a1;
  hW2[n * 64 + t]  = a2;

  if (t < CD) {
    float e1 = 0.f, e2 = 0.f;
    for (int f = 0; f < FD; ++f) {
      e1 += hn[f] * W_ew[f * CD + t];
      e2 += hn[f] * W_ew[(FD + f) * CD + t];
    }
    ewj[n * CD + t] = e1;
    ewi[n * CD + t] = e2;
  }
  if (t == 0) { float s = 0.f; for (int f = 0; f < FD; ++f) s += hn[f] * W_sa[f];      hsj[n] = s; }
  if (t == 1) { float s = 0.f; for (int f = 0; f < FD; ++f) s += hn[f] * W_sa[FD + f]; hsi[n] = s; }
}

// ---------------- kernel B: pack f16 B fragments ----------------
// 28 fragments total: df1 (2Kt x 4Nt), df2 (2x4), ew_he (2x2), cm1 (2x4)
// B layout (32x16 f16): lane l -> col = l%16, Kbase = (l/16)*8; dword w holds K=Kbase+2w(+8 if w>=4), +1
__global__ void sake_pack(const float* __restrict__ Wdf1, const float* __restrict__ Wdf2,
                          const float* __restrict__ Wewhe, const float* __restrict__ Wcm1,
                          uint32_t* __restrict__ pk_df1, uint32_t* __restrict__ pk_df2,
                          uint32_t* __restrict__ pk_ew,  uint32_t* __restrict__ pk_cm1) {
  int gid = blockIdx.x * blockDim.x + threadIdx.x;
  if (gid >= 28 * 32) return;
  int frag = gid >> 5, lane = gid & 31;

  const float* W; uint32_t* out; int Ncols, Kreal, ntiles, f;
  if (frag < 8)       { W = Wdf1;  out = pk_df1; Ncols = 64; Kreal = KR; ntiles = 4; f = frag; }
  else if (frag < 16) { W = Wdf2;  out = pk_df2; Ncols = 64; Kreal = 64; ntiles = 4; f = frag - 8; }
  else if (frag < 20) { W = Wewhe; out = pk_ew;  Ncols = 32; Kreal = 64; ntiles = 2; f = frag - 16; }
  else                { W = Wcm1;  out = pk_cm1; Ncols = 64; Kreal = 64; ntiles = 4; f = frag - 20; }

  int kt = f / ntiles, nt = f % ntiles;
  int col   = nt * 16 + (lane & 15);
  int kbase = kt * 32 + (lane >> 4) * 8;
#pragma unroll
  for (int w = 0; w < 8; ++w) {
    uint32_t d = 0;
#pragma unroll
    for (int hh = 0; hh < 2; ++hh) {
      int k = kbase + 2 * w + hh + ((w >= 4) ? 8 : 0);
      float v = (k < Kreal) ? W[k * Ncols + col] : 0.f;
      HU cv; cv.h = (_Float16)v;
      d |= ((uint32_t)cv.u) << (16 * hh);
    }
    out[((f * 32 + lane) << 3) + w] = d;
  }
}

// ---------------- block reduction helpers (512 elems, 256 threads) ----------------
__device__ __forceinline__ float blockMax512(const float* arr, float* red, int t) {
  float v = fmaxf(arr[t], arr[t + 256]);
  red[t] = v; __syncthreads();
  for (int s = 128; s > 0; s >>= 1) { if (t < s) red[t] = fmaxf(red[t], red[t + s]); __syncthreads(); }
  float r = red[0]; __syncthreads(); return r;
}
__device__ __forceinline__ float blockSumExp512(const float* arr, float m, float* red, int t) {
  float v = expf(arr[t] - m) + expf(arr[t + 256] - m);
  red[t] = v; __syncthreads();
  for (int s = 128; s > 0; s >>= 1) { if (t < s) red[t] += red[t + s]; __syncthreads(); }
  float r = red[0]; __syncthreads(); return r;
}

// ---------------- kernel C: fused per-node-i pairwise pipeline ----------------
// grid = 512 (one workgroup per i), block = 256 (8 waves); ~176KB dynamic LDS
__global__ __launch_bounds__(256, 1)
void sake_main(const float* __restrict__ x,    const float* __restrict__ h,
               const float* __restrict__ hW1b, const float* __restrict__ hW2,
               const float* __restrict__ hsj,  const float* __restrict__ hsi,
               const float* __restrict__ ewj,  const float* __restrict__ ewi,
               const uint32_t* __restrict__ pk_df1, const uint32_t* __restrict__ pk_df2,
               const uint32_t* __restrict__ pk_ew,  const uint32_t* __restrict__ pk_cm1,
               const float* __restrict__ b_df1, const float* __restrict__ b_df2,
               const float* __restrict__ b_ew,  const float* __restrict__ b_cm1,
               const float* __restrict__ W_cm2, const float* __restrict__ b_cm2,
               const float* __restrict__ W_pn1, const float* __restrict__ b_pn1,
               const float* __restrict__ W_pn2, const float* __restrict__ b_pn2,
               const float* __restrict__ W_nm1, const float* __restrict__ b_nm1,
               const float* __restrict__ W_nm2, const float* __restrict__ b_nm2,
               float* __restrict__ out) {
  extern __shared__ float smem[];
  float* he_s   = smem;                 // [512][64]
  float* stage  = smem + 32768;         // per-wave [16][64]
  float* xd     = smem + 40960;         // per-wave [16][4] (x_dir)
  float* dist_s = smem + 41472;         // [512]
  float* slog_s = smem + 41984;         // [512]
  float* cw_s   = smem + 42496;         // [512] coord_w partials
  float* red    = smem + 43008;         // [256]
  float* xatt   = smem + 43264;         // [32][3]
  float* hw2i   = smem + 43360;         // [64]
  float* ewi_s  = smem + 43424;         // [32]
  float* bdf1_s = smem + 43456;         // [64]
  float* bdf2_s = smem + 43520;         // [64]
  float* bew_s  = smem + 43584;         // [32]
  float* bcm1_s = smem + 43616;         // [64]
  float* wcm2_s = smem + 43680;         // [64]
  float* heagg  = smem + 43744;         // [64]
  float* xn_s   = smem + 43808;         // [32]
  float* pn1_s  = smem + 43840;         // [64]
  float* ne_s   = smem + 43904;         // [64]
  float* nm1_s  = smem + 43968;         // [64]
  float* xnew_s = smem + 44032;         // [4]
  float* misc   = smem + 44036;         // [0]=hsi_i [1..3]=x_i [4]=b_cm2

  const int i    = blockIdx.x;
  const int t    = threadIdx.x;
  const int wave = t >> 5;
  const int lane = t & 31;
  float* stw = stage + wave * 1024;     // 16x64
  float* xdw = xd + wave * 64;          // 16x4

  // ---- per-i setup ----
  if (t < 64) {
    hw2i[t]   = hW2[i * 64 + t];
    bdf1_s[t] = b_df1[t];
    bdf2_s[t] = b_df2[t];
    bcm1_s[t] = b_cm1[t];
    wcm2_s[t] = W_cm2[t];
  }
  if (t < 32) { ewi_s[t] = ewi[i * CD + t]; bew_s[t] = b_ew[t]; }
  if (t == 0) { misc[0] = hsi[i]; misc[4] = b_cm2[0]; }
  if (t < 3)  misc[1 + t] = x[i * 3 + t];
  cw_s[t] = 0.f; cw_s[t + 256] = 0.f;
  if (t < 96) xatt[t] = 0.f;
  if (t < 4)  xnew_s[t] = 0.f;
  __syncthreads();

  const float xi0 = misc[1], xi1 = misc[2], xi2 = misc[3];
  const float hsi_i = misc[0];
  const int   m     = lane & 15;
  const int   b0    = (lane & 16) ? 8 : 0;
  const int   rbase = (lane >> 4) << 3;

  float xacc[2][3] = {{0.f, 0.f, 0.f}, {0.f, 0.f, 0.f}};

  // ---- pass 1: per-wave j-tiles of 16 ----
  for (int tile = wave; tile < 32; tile += 8) {
    const int jbase = tile * 16;
    const int jrow  = jbase + m;

    // distance / semantic logit / x_dir for this tile
    float dxx = x[jrow * 3 + 0] - xi0;
    float dyy = x[jrow * 3 + 1] - xi1;
    float dzz = x[jrow * 3 + 2] - xi2;
    float d2   = dxx * dxx + dyy * dyy + dzz * dzz;
    float dist = sqrtf(fmaxf(d2, 0.f) + 1e-14f);
    if (lane < 16) {
      dist_s[jrow] = dist;
      float sa = hsj[jrow] + hsi_i;
      slog_s[jrow] = siluf(sa);
      float inv = 1.f / (d2 + 1e-5f);
      xdw[m * 4 + 0] = dxx * inv;
      xdw[m * 4 + 1] = dyy * inv;
      xdw[m * 4 + 2] = dzz * inv;
    }
    __builtin_amdgcn_wave_barrier();

    // A fragments for he0 = (hW1b[j] + hW2[i]) * rbf(dist), K padded 50->64
    H16 A0, A1;
    const float* hw1r = hW1b + jrow * 64;
#pragma unroll
    for (int e = 0; e < 16; ++e) {
      int k0 = a_koff(e, b0);
      {
        float mu = (float)k0 * (5.0f / 49.0f);
        float dd = dist - mu;
        float v  = (hw1r[k0] + hw2i[k0]) * expf(-10.f * dd * dd);
        A0.h[e] = (_Float16)v;
      }
      int k1 = 32 + k0;
      float v1 = 0.f;
      if (k1 < KR) {
        float mu = (float)k1 * (5.0f / 49.0f);
        float dd = dist - mu;
        v1 = (hw1r[k1] + hw2i[k1]) * expf(-10.f * dd * dd);
      }
      A1.h[e] = (_Float16)v1;
    }

    // GEMM1: he1 = silu(he0 @ W_df1 + b_df1) -> per-wave stage
#pragma unroll
    for (int nt = 0; nt < 4; ++nt) {
      v8f acc = {};
      acc = wmma_f16(A0.v, load_bfrag(pk_df1, nt, lane), acc);
      acc = wmma_f16(A1.v, load_bfrag(pk_df1, 4 + nt, lane), acc);
      int col = nt * 16 + m;
      float bias = bdf1_s[col];
#pragma unroll
      for (int r = 0; r < 8; ++r) stw[(rbase + r) * 64 + col] = siluf(acc[r] + bias);
    }
    __builtin_amdgcn_wave_barrier();

    // GEMM2: he = he1 @ W_df2 + b_df2 -> he_s (shared, reused in pass 2)
    H16 S0, S1;
#pragma unroll
    for (int e = 0; e < 16; ++e) {
      int k = a_koff(e, b0);
      S0.h[e] = (_Float16)stw[m * 64 + k];
      S1.h[e] = (_Float16)stw[m * 64 + 32 + k];
    }
#pragma unroll
    for (int nt = 0; nt < 4; ++nt) {
      v8f acc = {};
      acc = wmma_f16(S0.v, load_bfrag(pk_df2, nt, lane), acc);
      acc = wmma_f16(S1.v, load_bfrag(pk_df2, 4 + nt, lane), acc);
      int col = nt * 16 + m;
      float bias = bdf2_s[col];
#pragma unroll
      for (int r = 0; r < 8; ++r) he_s[(jbase + rbase + r) * 64 + col] = acc[r] + bias;
    }
    __builtin_amdgcn_wave_barrier();

    // A fragments from he
    H16 E0, E1;
#pragma unroll
    for (int e = 0; e < 16; ++e) {
      int k = a_koff(e, b0);
      E0.h[e] = (_Float16)he_s[jrow * 64 + k];
      E1.h[e] = (_Float16)he_s[jrow * 64 + 32 + k];
    }

    // GEMM3: w_edge = tanh(ewj + ewi + he @ W_ew_he + b_ew); accumulate x_att
#pragma unroll
    for (int nt = 0; nt < 2; ++nt) {
      v8f acc = {};
      acc = wmma_f16(E0.v, load_bfrag(pk_ew, nt, lane), acc);
      acc = wmma_f16(E1.v, load_bfrag(pk_ew, 2 + nt, lane), acc);
      int c = nt * 16 + m;
      float be = bew_s[c] + ewi_s[c];
#pragma unroll
      for (int r = 0; r < 8; ++r) {
        int row = rbase + r;
        int jj  = jbase + row;
        float w = tanhf(acc[r] + ewj[jj * CD + c] + be);
        xacc[nt][0] += w * xdw[row * 4 + 0];
        xacc[nt][1] += w * xdw[row * 4 + 1];
        xacc[nt][2] += w * xdw[row * 4 + 2];
      }
    }

    // GEMM4: coord_w = silu(he @ W_cm1 + b_cm1) . W_cm2  (bias added in pass 2)
    float p[8] = {0.f, 0.f, 0.f, 0.f, 0.f, 0.f, 0.f, 0.f};
#pragma unroll
    for (int nt = 0; nt < 4; ++nt) {
      v8f acc = {};
      acc = wmma_f16(E0.v, load_bfrag(pk_cm1, nt, lane), acc);
      acc = wmma_f16(E1.v, load_bfrag(pk_cm1, 4 + nt, lane), acc);
      int col = nt * 16 + m;
      float bias = bcm1_s[col], w2 = wcm2_s[col];
#pragma unroll
      for (int r = 0; r < 8; ++r) p[r] += siluf(acc[r] + bias) * w2;
    }
#pragma unroll
    for (int r = 0; r < 8; ++r) atomicAdd(&cw_s[jbase + rbase + r], p[r]);
  }

#pragma unroll
  for (int nt = 0; nt < 2; ++nt) {
    int c = nt * 16 + m;
    atomicAdd(&xatt[c * 3 + 0], xacc[nt][0]);
    atomicAdd(&xatt[c * 3 + 1], xacc[nt][1]);
    atomicAdd(&xatt[c * 3 + 2], xacc[nt][2]);
  }
  __syncthreads();

  // ---- pass 2: softmaxes, aggregation, node-level MLPs ----
  float md = blockMax512(dist_s, red, t);
  float Zd = blockSumExp512(dist_s, md, red, t);
  float ms = blockMax512(slog_s, red, t);
  float Zs = blockSumExp512(slog_s, ms, red, t);
  for (int j = t; j < NN; j += 256) {
    float tp = (expf(slog_s[j] - ms) / Zs) * (expf(dist_s[j] - md) / Zd);
    dist_s[j] = tp;   // semantic_att * spatial_att
  }
  __syncthreads();
  float mt = blockMax512(dist_s, red, t);
  float Zt = blockSumExp512(dist_s, mt, red, t);
  for (int j = t; j < NN; j += 256) slog_s[j] = expf(dist_s[j] - mt) / Zt;  // total_att
  __syncthreads();

  // he_agg = sum_j total_att[j] * he[j]
  {
    int f = t & 63, c4 = t >> 6;
    float acc = 0.f;
    for (int j = c4 * 128; j < c4 * 128 + 128; ++j) acc += slog_s[j] * he_s[j * 64 + f];
    red[t] = acc; __syncthreads();
    if (t < 64) heagg[t] = red[t] + red[64 + t] + red[128 + t] + red[192 + t];
    __syncthreads();
  }

  // x_att_norm -> norm_emb (pn MLP)
  if (t < 32) {
    float s = xatt[t * 3 + 0] * xatt[t * 3 + 0] +
              xatt[t * 3 + 1] * xatt[t * 3 + 1] +
              xatt[t * 3 + 2] * xatt[t * 3 + 2];
    xn_s[t] = sqrtf(fmaxf(s, 0.f) + 1e-14f);
  }
  __syncthreads();
  if (t < 64) {
    float a = b_pn1[t];
    for (int k = 0; k < 32; ++k) a += xn_s[k] * W_pn1[k * 64 + t];
    pn1_s[t] = siluf(a);
  }
  __syncthreads();
  if (t < 64) {
    float a = b_pn2[t];
    for (int k = 0; k < 64; ++k) a += pn1_s[k] * W_pn2[k * 64 + t];
    ne_s[t] = a;
  }
  __syncthreads();

  // x_new = sum_j (x[j]-x[i]) * coord_w[j] + x[i]
  {
    float bc = misc[4];
    float px = 0.f, py = 0.f, pz = 0.f;
    for (int j = t; j < NN; j += 256) {
      float w = cw_s[j] + bc;
      px += (x[j * 3 + 0] - xi0) * w;
      py += (x[j * 3 + 1] - xi1) * w;
      pz += (x[j * 3 + 2] - xi2) * w;
    }
    atomicAdd(&xnew_s[0], px);
    atomicAdd(&xnew_s[1], py);
    atomicAdd(&xnew_s[2], pz);
  }
  __syncthreads();
  if (t < 3) out[NN * FD + i * 3 + t] = xnew_s[t] + misc[1 + t];

  // node MLP: h_new = nm2(silu(nm1([h, he_agg, norm_emb])))
  if (t < 64) {
    float a = b_nm1[t];
    for (int k = 0; k < 64; ++k) a += h[i * 64 + k] * W_nm1[k * 64 + t];
    for (int k = 0; k < 64; ++k) a += heagg[k] * W_nm1[(64 + k) * 64 + t];
    for (int k = 0; k < 64; ++k) a += ne_s[k] * W_nm1[(128 + k) * 64 + t];
    nm1_s[t] = siluf(a);
  }
  __syncthreads();
  if (t < 64) {
    float a = b_nm2[t];
    for (int k = 0; k < 64; ++k) a += nm1_s[k] * W_nm2[k * 64 + t];
    out[i * 64 + t] = a;
  }
}

// ---------------- host launch ----------------
extern "C" void kernel_launch(void* const* d_in, const int* in_sizes, int n_in,
                              void* d_out, int out_size, void* d_ws, size_t ws_size,
                              hipStream_t stream) {
  (void)in_sizes; (void)n_in; (void)out_size; (void)ws_size;
  const float* h       = (const float*)d_in[0];
  const float* x       = (const float*)d_in[1];
  const float* W_df_in = (const float*)d_in[2];
  const float* b_df_in = (const float*)d_in[3];
  const float* W_df1   = (const float*)d_in[4];
  const float* b_df1   = (const float*)d_in[5];
  const float* W_df2   = (const float*)d_in[6];
  const float* b_df2   = (const float*)d_in[7];
  const float* W_ew    = (const float*)d_in[8];
  const float* b_ew    = (const float*)d_in[9];
  const float* W_pn1   = (const float*)d_in[10];
  const float* b_pn1   = (const float*)d_in[11];
  const float* W_pn2   = (const float*)d_in[12];
  const float* b_pn2   = (const float*)d_in[13];
  const float* W_nm1   = (const float*)d_in[14];
  const float* b_nm1   = (const float*)d_in[15];
  const float* W_nm2   = (const float*)d_in[16];
  const float* b_nm2   = (const float*)d_in[17];
  const float* W_cm1   = (const float*)d_in[18];
  const float* b_cm1   = (const float*)d_in[19];
  const float* W_cm2   = (const float*)d_in[20];
  const float* b_cm2   = (const float*)d_in[21];
  const float* W_sa    = (const float*)d_in[22];

  float* ws = (float*)d_ws;
  float* hW1b = ws;              // 512*64
  float* hW2  = ws + 32768;      // 512*64
  float* hsj  = ws + 65536;      // 512
  float* hsi  = ws + 66048;      // 512
  float* ewjp = ws + 66560;      // 512*32
  float* ewip = ws + 82944;      // 512*32
  uint32_t* pk     = (uint32_t*)(ws + 99328);
  uint32_t* pk_df1 = pk;           // 8 frags * 256
  uint32_t* pk_df2 = pk + 2048;    // 8 frags
  uint32_t* pk_ew  = pk + 4096;    // 4 frags
  uint32_t* pk_cm1 = pk + 5120;    // 8 frags

  sake_prep<<<NN, 64, 0, stream>>>(h, W_df_in, b_df_in, W_sa, W_ew,
                                   hW1b, hW2, hsj, hsi, ewjp, ewip);
  sake_pack<<<4, 256, 0, stream>>>(W_df1, W_df2, W_ew + 128 * CD, W_cm1,
                                   pk_df1, pk_df2, pk_ew, pk_cm1);

  const size_t smem_bytes = 44048u * sizeof(float);  // ~176 KB of the 320 KB WGP pool
  sake_main<<<NN, 256, smem_bytes, stream>>>(
      x, h, hW1b, hW2, hsj, hsi, ewjp, ewip,
      pk_df1, pk_df2, pk_ew, pk_cm1,
      b_df1, b_df2, b_ew, b_cm1, W_cm2, b_cm2,
      W_pn1, b_pn1, W_pn2, b_pn2, W_nm1, b_nm1, W_nm2, b_nm2,
      (float*)d_out);
}
